// PerformerBlock_33689723469904
// MI455X (gfx1250) — matrix operand order repaired and checked by reading, not compiled, not run on hardware
//
#include <hip/hip_runtime.h>
#include <hip/hip_bf16.h>

// ---------------------------------------------------------------------------
// Performer attention block for MI455X (gfx1250): all GEMMs via
// v_wmma_f32_16x16x32_bf16 (bf16 in, fp32 accumulate), B-tile staged in LDS
// through the gfx1250 async global->LDS path (ASYNCcnt) with double buffering.
// Shapes fixed by the reference: L=4096, N=4, E=1024, H=16, D=64.
// ---------------------------------------------------------------------------

typedef __bf16 bf16_t;
typedef __attribute__((ext_vector_type(16))) __bf16 v16bf;
typedef __attribute__((ext_vector_type(8)))  float  v8f;
typedef int v4i_t __attribute__((vector_size(16)));

#define LQ   4096
#define NBQ  4
#define EE   1024
#define HH   16
#define DD   64
#define NH   (NBQ * HH)      // 64 batches (n,h)
#define MROWS (LQ * NBQ)     // 16384 rows of the big GEMMs

// gfx1250 async global->LDS support (guarded; falls back to VGPR staging)
#if defined(__has_builtin)
#if __has_builtin(__builtin_amdgcn_global_load_async_to_lds_b128)
#define HAVE_ASYNC_LDS 1
#endif
#endif
#ifndef HAVE_ASYNC_LDS
#define HAVE_ASYNC_LDS 0
#endif

#if HAVE_ASYNC_LDS
typedef __attribute__((address_space(1))) v4i_t* gptr_v4i;   // global int4*
typedef __attribute__((address_space(3))) v4i_t* lptr_v4i;   // LDS int4*
#endif

template <int N>
__device__ __forceinline__ void wait_async_le() {
#if HAVE_ASYNC_LDS
#if defined(__has_builtin) && __has_builtin(__builtin_amdgcn_s_wait_asynccnt)
    __builtin_amdgcn_s_wait_asynccnt(N);
#else
    asm volatile("s_wait_asynccnt %0" ::"i"(N) : "memory");
#endif
#endif
}

// ---------------------------------------------------------------------------
// fp32 -> bf16 elementwise convert
// ---------------------------------------------------------------------------
__global__ void cvt_f32_bf16(const float* __restrict__ in, bf16_t* __restrict__ out, long n) {
    long i = (long)blockIdx.x * blockDim.x + threadIdx.x;
    if (i < n) out[i] = (bf16_t)in[i];
}

// ---------------------------------------------------------------------------
// Per-head modified Gram-Schmidt on the rows of projection_matrix[h] (64x64).
// MGS => positive R diag, so the reference's sign correction is the identity.
// Writes P^T in bf16, replicated per n so batch b = n*H + h uses plain strides.
// PtBf[b][e][d] = P[h][d][e]
// ---------------------------------------------------------------------------
__global__ void qr_orthonormalize(const float* __restrict__ projmat, bf16_t* __restrict__ PtBf) {
    __shared__ float R[DD * DD];
    __shared__ float invnrm;
    const int h = blockIdx.x;
    const int t = threadIdx.x;            // 64 threads
    for (int i = t; i < DD * DD; i += 64) R[i] = projmat[(long)h * DD * DD + i];
    __syncthreads();
    for (int i = 0; i < DD; ++i) {
        if (t == 0) {
            float s = 0.f;
            for (int d = 0; d < DD; ++d) s += R[i * DD + d] * R[i * DD + d];
            invnrm = rsqrtf(fmaxf(s, 1e-30f));
        }
        __syncthreads();
        R[i * DD + t] *= invnrm;
        __syncthreads();
        if (t > i) {
            float c = 0.f;
            for (int d = 0; d < DD; ++d) c += R[t * DD + d] * R[i * DD + d];
            for (int d = 0; d < DD; ++d) R[t * DD + d] -= c * R[i * DD + d];
        }
        __syncthreads();
    }
    for (int e = 0; e < DD; ++e) {
        bf16_t v = (bf16_t)R[t * DD + e];
        for (int n = 0; n < NBQ; ++n)
            PtBf[((long)(n * HH + h)) * DD * DD + (long)e * DD + t] = v;
    }
}

// ---------------------------------------------------------------------------
// Generic batched GEMM:  C[b][m][n] = (sum_k A[b][m][k] * B[b][n][k] + bias[n])
//                                     * outScale * (rowScale ? rowScale[b][m] : 1)
// A: bf16 row-major (lda, batch stride aStride); streamed straight from global
//    (no cross-wave reuse inside a block).
// B: bf16, N x K row-major (pre-transposed operand); the 64xK panel is shared
//    by all 8 waves of the block -> staged in LDS via async global->LDS loads,
//    double buffered in 64-wide K chunks.
// C: fp32 or bf16 (outBf16), ldc/cStride.
// One wave -> 32x64 tile = 2x4 v_wmma_f32_16x16x32_bf16; block = 8 waves in M.
// Requires M%32==0, N%64==0, K%64==0 (true for all uses here).
// ---------------------------------------------------------------------------
#define KCHUNK 64
#define BROW   72                  // padded LDS row (elements): 144B -> conflict-free b128 reads
#define BBUF   (64 * BROW)         // one buffer: 64 rows

__global__ void __launch_bounds__(256)
gemm_wmma_bf16(const bf16_t* __restrict__ A, long lda, long aStride,
               const bf16_t* __restrict__ B, long ldb, long bStride,
               void* __restrict__ Cv, long ldc, long cStride,
               int M, int N, int K,
               const float* __restrict__ bias,
               const float* __restrict__ rowScale, long rsStride,
               float outScale, int outBf16)
{
    __shared__ __align__(16) bf16_t smemB[2 * BBUF];   // 18 KB of the 320 KB WGP LDS

    const int tid  = threadIdx.x;
    const int lane = tid & 31;
    const int wave = tid >> 5;
    const int b    = blockIdx.z;
    const int m0   = blockIdx.x * 256 + wave * 32;
    const int n0   = blockIdx.y * 64;
    const bool active = (m0 < M);

    const bf16_t* Ab = A + (long)b * aStride;
    const bf16_t* Bb = B + (long)b * bStride;

    const int hl = lane & 15;
    const int hi = lane >> 4;

    // A fragment (16x32): lane<16 -> K=[0..7],[16..23]; lane>=16 -> K=[8..15],[24..31]
    const long aOff0 = (long)(m0 + hl) * lda + (hi ? 8 : 0);
    const long aOff1 = aOff0 + 16 * lda;

    // Cooperative B-panel fill: 64 rows x 64 k = 512 16B chunks, 2 per thread.
    auto fill = [&](int buf, int kc) {
#pragma unroll
        for (int ch = 0; ch < 2; ++ch) {
            const int chunk = tid + ch * 256;
            const int r = chunk >> 3;          // row 0..63
            const int c = chunk & 7;           // 16B chunk within row
            const bf16_t* g = Bb + (long)(n0 + r) * ldb + kc + c * 8;
            bf16_t* l = &smemB[buf * BBUF + r * BROW + c * 8];
#if HAVE_ASYNC_LDS
            __builtin_amdgcn_global_load_async_to_lds_b128((gptr_v4i)g, (lptr_v4i)l, 0, 0);
#else
            *(uint4*)l = *(const uint4*)g;
#endif
        }
    };

    v8f acc[2][4] = {};
    const int nk = K >> 6;                      // number of 64-wide K chunks
    fill(0, 0);

    for (int ck = 0; ck < nk; ++ck) {
        const int cur = ck & 1;
        if (ck + 1 < nk) {
            fill(cur ^ 1, (ck + 1) << 6);
            wait_async_le<2>();                 // this wave's chunk-ck fills done
        } else {
            wait_async_le<0>();
        }
        __syncthreads();                        // all waves' fills visible

        if (active) {
#pragma unroll
            for (int s = 0; s < 2; ++s) {       // two k=32 steps per chunk
                const int k0 = (ck << 6) + s * 32;
                v16bf afrag[2], bfrag[4];
                {
                    const bf16_t* p = Ab + aOff0 + k0;
                    ((uint4*)&afrag[0])[0] = *(const uint4*)(p);
                    ((uint4*)&afrag[0])[1] = *(const uint4*)(p + 16);
                    p = Ab + aOff1 + k0;
                    ((uint4*)&afrag[1])[0] = *(const uint4*)(p);
                    ((uint4*)&afrag[1])[1] = *(const uint4*)(p + 16);
                }
#pragma unroll
                for (int j = 0; j < 4; ++j) {
                    const bf16_t* lp = &smemB[cur * BBUF + (j * 16 + hl) * BROW + s * 32 + hi * 8];
                    ((uint4*)&bfrag[j])[0] = *(const uint4*)(lp);
                    ((uint4*)&bfrag[j])[1] = *(const uint4*)(lp + 16);
                }
#pragma unroll
                for (int i = 0; i < 2; ++i)
#pragma unroll
                    for (int j = 0; j < 4; ++j)
                        acc[i][j] = __builtin_amdgcn_wmma_f32_16x16x32_bf16(
                            false, afrag[i], false, bfrag[j], (short)0, acc[i][j], false, false);
            }
        }
        __syncthreads();                        // buffer may be refilled next iter
    }

    if (!active) return;

    // Epilogue. C/D layout: VGPR r holds M = (hi*8 + r), N = hl within the tile.
    const long cb = (long)b * cStride;
#pragma unroll
    for (int i = 0; i < 2; ++i) {
#pragma unroll
        for (int j = 0; j < 4; ++j) {
            const float* af = (const float*)&acc[i][j];
            const int n = n0 + j * 16 + hl;
            const float bb = bias ? bias[n] : 0.f;
#pragma unroll
            for (int r = 0; r < 8; ++r) {
                const int m = m0 + i * 16 + hi * 8 + r;
                float v = (af[r] + bb) * outScale;
                if (rowScale) v *= rowScale[(long)b * rsStride + m];
                const long off = cb + (long)m * ldc + n;
                if (outBf16) ((bf16_t*)Cv)[off] = (bf16_t)v;
                else         ((float*)Cv)[off]  = v;
            }
        }
    }
}

// ---------------------------------------------------------------------------
// Query feature map: per row (b,l): qp = exp(proj - rowmax(proj)) * D^-0.5 + eps
// (the 0.5*||x||^2 term cancels against the per-row max for queries)
// ---------------------------------------------------------------------------
__global__ void feature_q(const float* __restrict__ projQ, bf16_t* __restrict__ qp) {
    const long idx = (long)blockIdx.x * blockDim.x + threadIdx.x;  // b*L + l
    const float* p = projQ + idx * DD;
    float mx = -1e30f;
    for (int e = 0; e < DD; ++e) mx = fmaxf(mx, p[e]);
    bf16_t* o = qp + idx * DD;
    for (int e = 0; e < DD; ++e)
        o[e] = (bf16_t)(__expf(p[e] - mx) * 0.125f + 1e-6f);
}

__global__ void init_kmax(float* kmax) {
    if (threadIdx.x < NH) kmax[threadIdx.x] = -1e30f;
}

// Key pass 1: delta = proj - 0.5*||x||^2 (in place), atomic-max per (n,h) batch
__global__ void feature_k1(const bf16_t* __restrict__ ks, float* __restrict__ projK,
                           float* __restrict__ kmax) {
    __shared__ float sm[256];
    const int b = blockIdx.y;
    const int l = blockIdx.x * 256 + threadIdx.x;
    const bf16_t* x = ks + (long)l * (NBQ * EE) + (long)b * DD;
    float h = 0.f;
    for (int d = 0; d < DD; ++d) { float v = (float)x[d]; h += v * v; }
    h *= 0.5f;
    float* p = projK + ((long)b * LQ + l) * DD;
    float mx = -1e30f;
    for (int e = 0; e < DD; ++e) { float d2 = p[e] - h; p[e] = d2; mx = fmaxf(mx, d2); }
    sm[threadIdx.x] = mx;
    __syncthreads();
    for (int s = 128; s > 0; s >>= 1) {
        if (threadIdx.x < s) sm[threadIdx.x] = fmaxf(sm[threadIdx.x], sm[threadIdx.x + s]);
        __syncthreads();
    }
    if (threadIdx.x == 0) {
        const float val = sm[0];
        int* addr = (int*)(kmax + b);
        int old = atomicAdd(addr, 0);
        while (val > __int_as_float(old)) {
            int assumed = old;
            old = atomicCAS(addr, assumed, __float_as_int(val));
            if (old == assumed) break;
        }
    }
}

// Key pass 2: kp = exp(delta - kmax[b]) * D^-0.5 + eps, stored TRANSPOSED:
// kpT[b][d][l]  (contiguous in l -> K-contiguous operand for the kv GEMM)
__global__ void feature_k2(const float* __restrict__ projK, const float* __restrict__ kmax,
                           bf16_t* __restrict__ kpT) {
    const long idx = (long)blockIdx.x * blockDim.x + threadIdx.x;
    const int b = (int)(idx >> 12);
    const int l = (int)(idx & (LQ - 1));
    const float mx = kmax[b];
    const float* p = projK + idx * DD;
    bf16_t* o = kpT + (long)b * DD * LQ + l;
    for (int e = 0; e < DD; ++e)
        o[(long)e * LQ] = (bf16_t)(__expf(p[e] - mx) * 0.125f + 1e-6f);
}

// vsT[b][e][l] = vs[l][b*64 + e]   (head-split transpose of V features)
__global__ void transpose_v(const bf16_t* __restrict__ vs, bf16_t* __restrict__ vsT) {
    const long idx = (long)blockIdx.x * blockDim.x + threadIdx.x;
    const int b = (int)(idx >> 12);
    const int l = (int)(idx & (LQ - 1));
    const bf16_t* src = vs + (long)l * (NBQ * EE) + (long)b * DD;
    bf16_t* o = vsT + (long)b * DD * LQ + l;
    for (int e = 0; e < DD; ++e) o[(long)e * LQ] = src[e];
}

// ksum[b][d] = sum_l kpT[b][d][l]
__global__ void ksum_reduce(const bf16_t* __restrict__ kpT, float* __restrict__ ksum) {
    __shared__ float sm[256];
    const long row = blockIdx.x;                 // b*64 + d
    const bf16_t* p = kpT + row * LQ;
    float s = 0.f;
    for (int l = threadIdx.x; l < LQ; l += 256) s += (float)p[l];
    sm[threadIdx.x] = s;
    __syncthreads();
    for (int st = 128; st > 0; st >>= 1) {
        if (threadIdx.x < st) sm[threadIdx.x] += sm[threadIdx.x + st];
        __syncthreads();
    }
    if (threadIdx.x == 0) ksum[row] = sm[0];
}

// Dinv[b*L + l] = 1 / max(qp[b,l,:] . ksum[b,:], 1e-6)
__global__ void denom_kernel(const bf16_t* __restrict__ qp, const float* __restrict__ ksum,
                             float* __restrict__ Dinv) {
    const long idx = (long)blockIdx.x * blockDim.x + threadIdx.x;
    const int b = (int)(idx >> 12);
    const bf16_t* q = qp + idx * DD;
    const float* ks = ksum + (long)b * DD;
    float s = 0.f;
    for (int d = 0; d < DD; ++d) s += (float)q[d] * ks[d];
    Dinv[idx] = 1.f / fmaxf(s, 1e-6f);
}

// ---------------------------------------------------------------------------
extern "C" void kernel_launch(void* const* d_in, const int* in_sizes, int n_in,
                              void* d_out, int out_size, void* d_ws, size_t ws_size,
                              hipStream_t stream) {
    (void)in_sizes; (void)n_in; (void)out_size; (void)ws_size;
    const float* query = (const float*)d_in[0];
    const float* keyi  = (const float*)d_in[1];
    const float* value = (const float*)d_in[2];
    const float* ipw   = (const float*)d_in[3];   // (3E, E)
    const float* ipb   = (const float*)d_in[4];   // (3E)
    const float* opw   = (const float*)d_in[5];   // (E, E)
    const float* opb   = (const float*)d_in[6];   // (E)
    const float* pm    = (const float*)d_in[7];   // (H, D, D)

    // --- workspace carve ---
    char* w = (char*)d_ws;
    auto alloc = [&](size_t bytes) -> char* {
        char* p = w;
        w += (bytes + 255) & ~(size_t)255;
        return p;
    };
    const long nX  = (long)MROWS * EE;            // 16M elements
    bf16_t* Xq    = (bf16_t*)alloc(nX * 2);
    bf16_t* Xk    = (bf16_t*)alloc(nX * 2);
    bf16_t* Xv    = (bf16_t*)alloc(nX * 2);
    bf16_t* Wbf   = (bf16_t*)alloc((long)3 * EE * EE * 2);
    bf16_t* Obf   = (bf16_t*)alloc((long)EE * EE * 2);
    bf16_t* PtBf  = (bf16_t*)alloc((long)NH * DD * DD * 2);
    bf16_t* qs    = (bf16_t*)alloc(nX * 2);       // scaled q features (bf16)
    bf16_t* ks    = (bf16_t*)alloc(nX * 2);
    bf16_t* vs    = (bf16_t*)alloc(nX * 2);
    float*  projQ = (float*)alloc((long)NH * LQ * DD * 4);
    float*  projK = (float*)alloc((long)NH * LQ * DD * 4);
    bf16_t* qp    = (bf16_t*)alloc((long)NH * LQ * DD * 2);
    bf16_t* kpT   = (bf16_t*)alloc((long)NH * DD * LQ * 2);
    bf16_t* vsT   = (bf16_t*)alloc((long)NH * DD * LQ * 2);
    bf16_t* kvT   = (bf16_t*)alloc((long)NH * DD * DD * 2);
    float*  kmax  = (float*)alloc(NH * 4);
    float*  ksum  = (float*)alloc((long)NH * DD * 4);
    float*  Dinv  = (float*)alloc((long)NH * LQ * 4);
    bf16_t* attn  = (bf16_t*)alloc(nX * 2);

    const float qscale = 0.35355339059327373f;    // D^-0.25, D=64

    // 1) convert inputs + weights to bf16
    cvt_f32_bf16<<<(unsigned)((nX + 255) / 256), 256, 0, stream>>>(query, Xq, nX);
    cvt_f32_bf16<<<(unsigned)((nX + 255) / 256), 256, 0, stream>>>(keyi,  Xk, nX);
    cvt_f32_bf16<<<(unsigned)((nX + 255) / 256), 256, 0, stream>>>(value, Xv, nX);
    {
        long nW = (long)3 * EE * EE;
        cvt_f32_bf16<<<(unsigned)((nW + 255) / 256), 256, 0, stream>>>(ipw, Wbf, nW);
        long nO = (long)EE * EE;
        cvt_f32_bf16<<<(unsigned)((nO + 255) / 256), 256, 0, stream>>>(opw, Obf, nO);
    }

    // 2) orthonormalize projection matrices -> P^T (bf16)
    qr_orthonormalize<<<HH, 64, 0, stream>>>(pm, PtBf);

    // 3) QKV in-projections (q,k scaled by D^-0.25 in epilogue)
    {
        dim3 g(MROWS / 256, EE / 64, 1);
        gemm_wmma_bf16<<<g, 256, 0, stream>>>(Xq, EE, 0, Wbf + 0L * EE * EE, EE, 0,
                                              qs, EE, 0, MROWS, EE, EE,
                                              ipb + 0 * EE, nullptr, 0, qscale, 1);
        gemm_wmma_bf16<<<g, 256, 0, stream>>>(Xk, EE, 0, Wbf + 1L * EE * EE, EE, 0,
                                              ks, EE, 0, MROWS, EE, EE,
                                              ipb + 1 * EE, nullptr, 0, qscale, 1);
        gemm_wmma_bf16<<<g, 256, 0, stream>>>(Xv, EE, 0, Wbf + 2L * EE * EE, EE, 0,
                                              vs, EE, 0, MROWS, EE, EE,
                                              ipb + 2 * EE, nullptr, 0, 1.f, 1);
    }

    // 4) feature projections: proj[b] = x[b] @ P[h]   (batch b = n*H + h)
    {
        dim3 g(LQ / 256, 1, NH);
        gemm_wmma_bf16<<<g, 256, 0, stream>>>(qs, NBQ * EE, DD, PtBf, DD, (long)DD * DD,
                                              projQ, DD, (long)LQ * DD, LQ, DD, DD,
                                              nullptr, nullptr, 0, 1.f, 0);
        gemm_wmma_bf16<<<g, 256, 0, stream>>>(ks, NBQ * EE, DD, PtBf, DD, (long)DD * DD,
                                              projK, DD, (long)LQ * DD, LQ, DD, DD,
                                              nullptr, nullptr, 0, 1.f, 0);
    }

    // 5) feature maps
    feature_q<<<(NH * LQ) / 256, 256, 0, stream>>>(projQ, qp);
    init_kmax<<<1, 64, 0, stream>>>(kmax);
    feature_k1<<<dim3(LQ / 256, NH), 256, 0, stream>>>(ks, projK, kmax);
    feature_k2<<<(NH * LQ) / 256, 256, 0, stream>>>(projK, kmax, kpT);
    transpose_v<<<(NH * LQ) / 256, 256, 0, stream>>>(vs, vsT);

    // 6) kv^T[b][e][d] = sum_l v[l,e] * kp[l,d]
    {
        dim3 g(1, 1, NH);
        gemm_wmma_bf16<<<g, 256, 0, stream>>>(vsT, LQ, (long)DD * LQ, kpT, LQ, (long)DD * LQ,
                                              kvT, DD, (long)DD * DD, DD, DD, LQ,
                                              nullptr, nullptr, 0, 1.f, 1);
    }

    // 7) denominator
    ksum_reduce<<<NH * DD, 256, 0, stream>>>(kpT, ksum);
    denom_kernel<<<(NH * LQ) / 256, 256, 0, stream>>>(qp, ksum, Dinv);

    // 8) out2[b][l][e] = Dinv[b][l] * (qp[b] @ kv[b]) -> packed into attn (L*N, E) bf16
    {
        dim3 g(LQ / 256, 1, NH);
        gemm_wmma_bf16<<<g, 256, 0, stream>>>(qp, DD, (long)LQ * DD, kvT, DD, (long)DD * DD,
                                              attn, NBQ * EE, DD, LQ, DD, DD,
                                              nullptr, Dinv, LQ, 1.f, 1);
    }

    // 9) output projection -> d_out (fp32)
    {
        dim3 g(MROWS / 256, EE / 64, 1);
        gemm_wmma_bf16<<<g, 256, 0, stream>>>(attn, EE, 0, Obf, EE, 0,
                                              d_out, EE, 0, MROWS, EE, EE,
                                              opb, nullptr, 0, 1.f, 0);
    }
}
